// GCN_52501680226728
// MI455X (gfx1250) — compile-verified, matrix operand order checked by reading, not run on hardware
//
#include <hip/hip_runtime.h>

typedef __attribute__((ext_vector_type(16))) _Float16 v16h;
typedef __attribute__((ext_vector_type(8)))  _Float16 v8h;
typedef __attribute__((ext_vector_type(8)))  float    v8f;

#define HD 128  // feature dim (D == H == 128)

// ---------------- weight prep: Wt[n][k] = (f16) W[k][n] ----------------
__global__ void gcn_prep_weights(const float* __restrict__ W1,
                                 const float* __restrict__ W2,
                                 _Float16* __restrict__ Wt1,
                                 _Float16* __restrict__ Wt2) {
  int n = blockIdx.x;   // output column
  int k = threadIdx.x;  // input row
  if (blockIdx.y == 0) Wt1[n * HD + k] = (_Float16)W1[k * HD + n];
  else                 Wt2[n * HD + k] = (_Float16)W2[k * HD + n];
}

// ---------------- degree / norm computation ----------------
__global__ void gcn_init_deg(unsigned* __restrict__ deg_out,
                             unsigned* __restrict__ deg_in, int N) {
  int i = blockIdx.x * blockDim.x + threadIdx.x;
  if (i < N) { deg_out[i] = 1u; deg_in[i] = 1u; }  // self-loop
}

__global__ void gcn_edge_deg(const int* __restrict__ src, const int* __restrict__ dst,
                             unsigned* __restrict__ deg_out, unsigned* __restrict__ deg_in,
                             int E) {
  int e = blockIdx.x * blockDim.x + threadIdx.x;
  if (e < E) {
    atomicAdd(&deg_out[src[e]], 1u);
    atomicAdd(&deg_in[dst[e]], 1u);
  }
}

__global__ void gcn_norms(const unsigned* __restrict__ deg_out,
                          const unsigned* __restrict__ deg_in,
                          float* __restrict__ norm_out, float* __restrict__ norm_in, int N) {
  int i = blockIdx.x * blockDim.x + threadIdx.x;
  if (i < N) {
    norm_out[i] = rsqrtf((float)deg_out[i]);  // deg >= 1 always (self-loop)
    norm_in[i]  = rsqrtf((float)deg_in[i]);
  }
}

// ---------------- agg init: self-loop term agg[i] = h[i]*norm_out[i] ----------------
__global__ void gcn_agg_init(const float* __restrict__ h, const float* __restrict__ norm_out,
                             float* __restrict__ agg, int total) {
  int i = blockIdx.x * blockDim.x + threadIdx.x;
  if (i < total) agg[i] = h[i] * norm_out[i >> 7];
}

// ---------------- edge scatter: one wave32 per edge, float4 per lane ----------------
// Edge metadata is wave-uniform: force scalar (SMEM) loads via readfirstlane.
// Accumulation: native global_atomic_add_f32 (relaxed, agent scope) into the
// L2-resident agg buffer.
__global__ void __launch_bounds__(256)
gcn_scatter(const float* __restrict__ h, const float* __restrict__ norm_out,
            const int* __restrict__ src, const int* __restrict__ dst,
            float* __restrict__ agg, int E) {
  int wid  = (int)((blockIdx.x * (unsigned)blockDim.x + threadIdx.x) >> 5);
  int lane = threadIdx.x & 31;
  if (wid >= E) return;
  int e = __builtin_amdgcn_readfirstlane(wid);   // wave-uniform edge id
  int s = src[e];
  int d = dst[e];
  float ns = norm_out[s];
  float4 v = ((const float4*)(h + (size_t)s * HD))[lane];  // 32 lanes x 16B = 512B row
  float* ap = agg + (size_t)d * HD + lane * 4;
  __hip_atomic_fetch_add(ap + 0, v.x * ns, __ATOMIC_RELAXED, __HIP_MEMORY_SCOPE_AGENT);
  __hip_atomic_fetch_add(ap + 1, v.y * ns, __ATOMIC_RELAXED, __HIP_MEMORY_SCOPE_AGENT);
  __hip_atomic_fetch_add(ap + 2, v.z * ns, __ATOMIC_RELAXED, __HIP_MEMORY_SCOPE_AGENT);
  __hip_atomic_fetch_add(ap + 3, v.w * ns, __ATOMIC_RELAXED, __HIP_MEMORY_SCOPE_AGENT);
}

// B-fragment loader: column (colBase+m), K segment per ISA lane-group split.
__device__ __forceinline__ v16h gcn_load_bfrag(const _Float16* __restrict__ bp,
                                               int kb, int g) {
  v8h b0 = *(const v8h*)(bp + kb + g * 8);
  v8h b1 = *(const v8h*)(bp + kb + 16 + g * 8);
  v16h b;
#pragma unroll
  for (int i = 0; i < 8; ++i) { b[i] = b0[i]; b[8 + i] = b1[i]; }
  return b;
}

// ---------------- WMMA GEMM: out = (agg * norm_in) @ W + b ----------------
// block = 256 threads = 8 waves; each WAVE owns one 16-row tile: A fragments
// (full K=128) loaded once into registers, then loop over the 8 column tiles
// with DOUBLE-BUFFERED B fragments so tile ct+1's weight loads overlap tile
// ct's 4 WMMAs (breaks the load->wait(0)->wmma serialization seen when B
// reuses one register set). 32 WMMAs per wave. Row-tile early-exit is
// wave-uniform, so EXEC is all-ones at every WMMA.
__global__ void __launch_bounds__(256)
gcn_gemm(const float* __restrict__ A, const float* __restrict__ norm_in,
         const _Float16* __restrict__ Wt,  // [HD][HD] = W transposed, f16
         const float* __restrict__ bias,
         float* __restrict__ out, int rowTiles) {
  const int lane = threadIdx.x & 31;
  const int wv   = threadIdx.x >> 5;
  const int rowTile = blockIdx.x * 8 + wv;
  if (rowTile >= rowTiles) return;

  const int g = lane >> 4;   // lane group (ISA K-split)
  const int m = lane & 15;
  const int arow = rowTile * 16 + m;
  const float s  = norm_in[arow];
  const float* ap = A + (size_t)arow * HD;

  // Load all 4 A fragments (full K=128 row) once; fuse norm_in scale + f16 cvt.
  v16h afrag[4];
#pragma unroll
  for (int kk = 0; kk < 4; ++kk) {
    const int kb = kk * 32;
    float4 f0 = *(const float4*)(ap + kb + g * 8);
    float4 f1 = *(const float4*)(ap + kb + g * 8 + 4);
    float4 f2 = *(const float4*)(ap + kb + 16 + g * 8);
    float4 f3 = *(const float4*)(ap + kb + 16 + g * 8 + 4);
    v16h a;
    a[0]  = (_Float16)(f0.x * s); a[1]  = (_Float16)(f0.y * s);
    a[2]  = (_Float16)(f0.z * s); a[3]  = (_Float16)(f0.w * s);
    a[4]  = (_Float16)(f1.x * s); a[5]  = (_Float16)(f1.y * s);
    a[6]  = (_Float16)(f1.z * s); a[7]  = (_Float16)(f1.w * s);
    a[8]  = (_Float16)(f2.x * s); a[9]  = (_Float16)(f2.y * s);
    a[10] = (_Float16)(f2.z * s); a[11] = (_Float16)(f2.w * s);
    a[12] = (_Float16)(f3.x * s); a[13] = (_Float16)(f3.y * s);
    a[14] = (_Float16)(f3.z * s); a[15] = (_Float16)(f3.w * s);
    afrag[kk] = a;
  }

  // Double-buffered B fragments: prefetch column tile 0.
  v16h bfr[2][4];
  {
    const _Float16* bp0 = Wt + (size_t)m * HD;
#pragma unroll
    for (int kk = 0; kk < 4; ++kk) bfr[0][kk] = gcn_load_bfrag(bp0, kk * 32, g);
  }

#pragma unroll
  for (int ct = 0; ct < 8; ++ct) {
    const int cur = ct & 1;
    // Prefetch next column tile's B fragments (overlaps this tile's WMMAs).
    if (ct < 7) {
      const _Float16* bpn = Wt + (size_t)((ct + 1) * 16 + m) * HD;
#pragma unroll
      for (int kk = 0; kk < 4; ++kk) bfr[cur ^ 1][kk] = gcn_load_bfrag(bpn, kk * 32, g);
    }

    // C/D layout: VGPR r, lane -> (M = r + g*8, N = m). Bias depends on N only.
    float bv = bias[ct * 16 + m];
    v8f c;
#pragma unroll
    for (int r = 0; r < 8; ++r) c[r] = bv;

#pragma unroll
    for (int kk = 0; kk < 4; ++kk) {
      c = __builtin_amdgcn_wmma_f32_16x16x32_f16(
          /*neg_a=*/false, afrag[kk], /*neg_b=*/false, bfr[cur][kk],
          /*c_mod=*/(short)0, c, /*reuse_a=*/false, /*reuse_b=*/false);
    }

    // Store: row = rowTile*16 + r + g*8, col = ct*16 + m
    float* op = out + (size_t)(rowTile * 16 + g * 8) * HD + ct * 16 + m;
#pragma unroll
    for (int r = 0; r < 8; ++r) op[(size_t)r * HD] = c[r];
  }
}

// ---------------- host side ----------------
extern "C" void kernel_launch(void* const* d_in, const int* in_sizes, int n_in,
                              void* d_out, int out_size, void* d_ws, size_t ws_size,
                              hipStream_t stream) {
  const float* x  = (const float*)d_in[0];
  const int*  src = (const int*)d_in[1];
  const int*  dst = (const int*)d_in[2];
  const float* W1 = (const float*)d_in[3];
  const float* b1 = (const float*)d_in[4];
  const float* W2 = (const float*)d_in[5];
  const float* b2 = (const float*)d_in[6];
  float* out = (float*)d_out;

  const int N = in_sizes[0] / HD;
  const int E = in_sizes[1];

  // workspace carve-up (~53 MB total; agg is L2-resident on MI455X)
  char* p = (char*)d_ws;
  float*    agg      = (float*)p;    p += (size_t)N * HD * sizeof(float);
  float*    norm_out = (float*)p;    p += (size_t)N * sizeof(float);
  float*    norm_in  = (float*)p;    p += (size_t)N * sizeof(float);
  unsigned* deg_out  = (unsigned*)p; p += (size_t)N * sizeof(unsigned);
  unsigned* deg_in   = (unsigned*)p; p += (size_t)N * sizeof(unsigned);
  _Float16* Wt1      = (_Float16*)p; p += (size_t)HD * HD * sizeof(_Float16);
  _Float16* Wt2      = (_Float16*)p;

  const int nb = (N + 255) / 256;
  const int eb = (E + 255) / 256;
  const int total = N * HD;
  const int tb = (total + 255) / 256;
  const int sb = (E + 7) / 8;            // 8 edges (waves) per 256-thread block
  const int rowTiles = N / 16;           // 6250 (N % 16 == 0)
  const int gb = (rowTiles + 7) / 8;     // 8 row tiles (waves) per block

  gcn_prep_weights<<<dim3(HD, 2), HD, 0, stream>>>(W1, W2, Wt1, Wt2);
  gcn_init_deg<<<nb, 256, 0, stream>>>(deg_out, deg_in, N);
  gcn_edge_deg<<<eb, 256, 0, stream>>>(src, dst, deg_out, deg_in, E);
  gcn_norms<<<nb, 256, 0, stream>>>(deg_out, deg_in, norm_out, norm_in, N);

  // ---- layer 1: h1 = (Â x) W1 + b1, h1 stored in d_out ----
  gcn_agg_init<<<tb, 256, 0, stream>>>(x, norm_out, agg, total);
  gcn_scatter<<<sb, 256, 0, stream>>>(x, norm_out, src, dst, agg, E);
  gcn_gemm<<<gb, 256, 0, stream>>>(agg, norm_in, Wt1, b1, out, rowTiles);

  // ---- layer 2: out = (Â h1) W2 + b2 ----
  gcn_agg_init<<<tb, 256, 0, stream>>>(out, norm_out, agg, total);
  gcn_scatter<<<sb, 256, 0, stream>>>(out, norm_out, src, dst, agg, E);
  gcn_gemm<<<gb, 256, 0, stream>>>(agg, norm_in, Wt2, b2, out, rowTiles);
}
